// TransformerBlock_54365696032814
// MI455X (gfx1250) — matrix operand order repaired and checked by reading, not compiled, not run on hardware
//
#include <hip/hip_runtime.h>
#include <hip/hip_bf16.h>

// ---------------- types / constants ----------------
typedef _Float16 v16h __attribute__((ext_vector_type(16)));
typedef _Float16 h8   __attribute__((ext_vector_type(8)));
typedef float    v8f  __attribute__((ext_vector_type(8)));

union Frag16 { v16h v; h8 h[2]; };

__device__ __forceinline__ v8f wmma_f16(v16h a, v16h b, v8f c) {
  // D = A(16x32 f16) * B(32x16 f16) + C(16x16 f32)
  return __builtin_amdgcn_wmma_f32_16x16x32_f16(false, a, false, b, (short)0, c, false, false);
}

// async global->LDS copy of 16 bytes per lane (gfx1250 ASYNCcnt path).
// Generic pointers to LDS carry the LDS byte offset in their low 32 bits.
__device__ __forceinline__ void async_copy_b128(void* lds_ptr, const void* gptr) {
  unsigned lds = (unsigned)(uintptr_t)lds_ptr;
  asm volatile("global_load_async_to_lds_b128 %0, %1, off"
               :: "v"(lds), "v"(gptr) : "memory");
}

#define B_    2
#define S_    2048
#define DM_   1024
#define NH_   16
#define DH_   64
#define DMLP_ 4096
#define ROWS_ (B_*S_)   // 4096

// ---------------- weight packing ----------------
__global__ void pack_qkv_w_kernel(const float* __restrict__ WQ, const float* __restrict__ WK,
                                  const float* __restrict__ WV, _Float16* __restrict__ out) {
  int id = blockIdx.x * 256 + threadIdx.x;           // over DM_ * 3*DM_
  if (id >= DM_ * 3 * DM_) return;
  int col = id % (3 * DM_);
  int m   = id / (3 * DM_);
  int which = col / DM_;
  int rem = col % DM_;
  int i = rem / DH_, h = rem % DH_;
  const float* W = (which == 0) ? WQ : (which == 1 ? WK : WV);
  out[id] = (_Float16)W[((size_t)i * DM_ + m) * DH_ + h];
}

__global__ void pack_qkv_b_kernel(const float* __restrict__ bQ, const float* __restrict__ bK,
                                  const float* __restrict__ bV, float* __restrict__ out) {
  int id = blockIdx.x * 256 + threadIdx.x;
  if (id >= 3 * DM_) return;
  int which = id / DM_, rem = id % DM_;
  const float* b = (which == 0) ? bQ : (which == 1 ? bK : bV);
  out[id] = b[rem];
}

__global__ void cvt_f16_kernel(const float* __restrict__ src, _Float16* __restrict__ dst, int n) {
  int id = blockIdx.x * 256 + threadIdx.x;
  if (id < n) dst[id] = (_Float16)src[id];
}

// vt[(b*NH+h)*DH + d][s] = qkv[(b*S+s)*3072 + 2048 + h*DH + d]
__global__ void transpose_v_kernel(const _Float16* __restrict__ qkv, _Float16* __restrict__ vt) {
  int id = blockIdx.x * 256 + threadIdx.x;           // B*NH*DH*S
  int s = id % S_;
  int d = (id / S_) % DH_;
  int h = (id / (S_ * DH_)) % NH_;
  int b =  id / (S_ * DH_ * NH_);
  vt[id] = qkv[((size_t)(b * S_ + s)) * (3 * DM_) + 2 * DM_ + h * DH_ + d];
}

// ---------------- layernorm (f32 in -> f16 out) ----------------
__global__ void ln_kernel(const float* __restrict__ x, const float* __restrict__ w,
                          const float* __restrict__ bb, _Float16* __restrict__ out, int D) {
  int row = blockIdx.x, tid = threadIdx.x;
  const float* xr = x + (size_t)row * D;
  __shared__ float red[256];
  float s = 0.f;
  for (int i = tid; i < D; i += 256) s += xr[i];
  red[tid] = s; __syncthreads();
  for (int o = 128; o > 0; o >>= 1) { if (tid < o) red[tid] += red[tid + o]; __syncthreads(); }
  float mean = red[0] / D; __syncthreads();
  s = 0.f;
  for (int i = tid; i < D; i += 256) { float d = xr[i] - mean; s += d * d; }
  red[tid] = s; __syncthreads();
  for (int o = 128; o > 0; o >>= 1) { if (tid < o) red[tid] += red[tid + o]; __syncthreads(); }
  float rstd = rsqrtf(red[0] / D + 1e-5f);
  _Float16* orow = out + (size_t)row * D;
  for (int i = tid; i < D; i += 256)
    orow[i] = (_Float16)((xr[i] - mean) * rstd * w[i] + bb[i]);
}

// ---------------- SoLU + LN: post = LN(softmax(pre)*pre) -> f16 ----------------
__global__ void solu_kernel(float* __restrict__ pre, const float* __restrict__ w,
                            const float* __restrict__ bb, _Float16* __restrict__ out, int D) {
  int row = blockIdx.x, tid = threadIdx.x;
  float* xr = pre + (size_t)row * D;
  __shared__ float red[256];
  float m = -1e30f;
  for (int i = tid; i < D; i += 256) m = fmaxf(m, xr[i]);
  red[tid] = m; __syncthreads();
  for (int o = 128; o > 0; o >>= 1) { if (tid < o) red[tid] = fmaxf(red[tid], red[tid + o]); __syncthreads(); }
  m = red[0]; __syncthreads();
  float se = 0.f;
  for (int i = tid; i < D; i += 256) se += __expf(xr[i] - m);
  red[tid] = se; __syncthreads();
  for (int o = 128; o > 0; o >>= 1) { if (tid < o) red[tid] += red[tid + o]; __syncthreads(); }
  float inv = 1.f / red[0]; __syncthreads();
  float s1 = 0.f, s2 = 0.f;
  for (int i = tid; i < D; i += 256) {
    float p = __expf(xr[i] - m) * inv * xr[i];
    xr[i] = p; s1 += p; s2 += p * p;
  }
  red[tid] = s1; __syncthreads();
  for (int o = 128; o > 0; o >>= 1) { if (tid < o) red[tid] += red[tid + o]; __syncthreads(); }
  s1 = red[0]; __syncthreads();
  red[tid] = s2; __syncthreads();
  for (int o = 128; o > 0; o >>= 1) { if (tid < o) red[tid] += red[tid + o]; __syncthreads(); }
  s2 = red[0];
  float mean = s1 / D;
  float rstd = rsqrtf(s2 / D - mean * mean + 1e-5f);
  _Float16* orow = out + (size_t)row * D;
  for (int i = tid; i < D; i += 256)
    orow[i] = (_Float16)((xr[i] - mean) * rstd * w[i] + bb[i]);
}

// ---------------- WMMA GEMM: C[M,N] = A[M,K]*B[K,N] (+bias +resid) ----------------
// block = 256 threads = 8 waves; block tile 128x64; wave grid 4(m) x 2(n);
// wave tile 32x32 -> 4 WMMAs per 32-deep K-chunk.
// Double-buffered LDS; A staged via async-to-LDS with in-order partial wait
// (s_wait_asynccnt 2 drains only the current tile while next tile is in flight).
__global__ __launch_bounds__(256) void gemm_f16_kernel(
    const _Float16* __restrict__ A, int lda,
    const _Float16* __restrict__ Bm, int ldb,
    const float* __restrict__ bias,
    const float* __restrict__ resid,
    float* __restrict__ outF, _Float16* __restrict__ outH,
    int ldc, int K)
{
  __shared__ __align__(16) _Float16 As[2][128][40];
  __shared__ __align__(16) _Float16 Bt[2][64][40];   // transposed: Bt[buf][n][k]
  const int tid = threadIdx.x;
  const int wave = tid >> 5, lane = tid & 31;
  const int lane16 = lane & 15, g = lane >> 4;
  const int wm = wave & 3;        // 0..3 -> 32-row slice
  const int wn = wave >> 2;       // 0..1 -> 32-col slice
  const int rowBlk = blockIdx.y * 128;
  const int colBlk = blockIdx.x * 64;

  v8f acc00 = {}, acc01 = {}, acc10 = {}, acc11 = {};

  // per-thread staging coordinates
  const int ar0 = tid >> 2,         ap0 = tid & 3;   // A chunk 0 (rows 0..63)
  const int ar1 = (tid + 256) >> 2, ap1 = tid & 3;   // A chunk 1 (rows 64..127)
  const int bkr = tid >> 3;                          // B row (k) 0..31
  const int bnq = (tid & 7) * 8;                     // B col group 0..56

  const _Float16* aptr0 = A + (size_t)(rowBlk + ar0) * lda + ap0 * 8;
  const _Float16* aptr1 = A + (size_t)(rowBlk + ar1) * lda + ap1 * 8;
  const _Float16* bptr  = Bm + (size_t)bkr * ldb + colBlk + bnq;
  const size_t bstep = (size_t)32 * ldb;

  // ---- prologue: stage tile 0 into buffer 0 ----
  async_copy_b128(&As[0][ar0][ap0 * 8], aptr0);
  async_copy_b128(&As[0][ar1][ap1 * 8], aptr1);
  {
    h8 bv = *(const h8*)bptr;
#pragma unroll
    for (int j2 = 0; j2 < 8; ++j2) Bt[0][bnq + j2][bkr] = bv[j2];
  }
  const _Float16* bnext = bptr + bstep;

  for (int k0 = 0; k0 < K; k0 += 32) {
    const int cur = (k0 >> 5) & 1;
    const int nxt = cur ^ 1;
    if (k0 + 32 < K) {
      // issue next tile while current tile finishes (async completes in order)
      async_copy_b128(&As[nxt][ar0][ap0 * 8], aptr0 + k0 + 32);
      async_copy_b128(&As[nxt][ar1][ap1 * 8], aptr1 + k0 + 32);
      {
        h8 bv = *(const h8*)bnext;
#pragma unroll
        for (int j2 = 0; j2 < 8; ++j2) Bt[nxt][bnq + j2][bkr] = bv[j2];
      }
      bnext += bstep;
      if (k0 + 64 < K) {
        __builtin_prefetch((const void*)(aptr0 + k0 + 64), 0, 1);
        __builtin_prefetch((const void*)(bnext), 0, 1);
      }
      asm volatile("s_wait_asynccnt 2" ::: "memory");  // drain current tile only
    } else {
      asm volatile("s_wait_asynccnt 0" ::: "memory");
    }
    __syncthreads();

    Frag16 Af0, Af1;
    {
      int r0 = wm * 32 + lane16;
      Af0.h[0] = *(const h8*)&As[cur][r0][16 * g];
      Af0.h[1] = *(const h8*)&As[cur][r0][16 * g + 8];
      Af1.h[0] = *(const h8*)&As[cur][r0 + 16][16 * g];
      Af1.h[1] = *(const h8*)&As[cur][r0 + 16][16 * g + 8];
    }
    Frag16 Bf0, Bf1;
    {
      int n0 = wn * 32 + lane16;
      Bf0.h[0] = *(const h8*)&Bt[cur][n0][16 * g];
      Bf0.h[1] = *(const h8*)&Bt[cur][n0][16 * g + 8];
      Bf1.h[0] = *(const h8*)&Bt[cur][n0 + 16][16 * g];
      Bf1.h[1] = *(const h8*)&Bt[cur][n0 + 16][16 * g + 8];
    }
    acc00 = wmma_f16(Af0.v, Bf0.v, acc00);
    acc01 = wmma_f16(Af0.v, Bf1.v, acc01);
    acc10 = wmma_f16(Af1.v, Bf0.v, acc10);
    acc11 = wmma_f16(Af1.v, Bf1.v, acc11);
    __syncthreads();   // readers done before buffer `cur` is restaged next+1
  }

  // epilogue: C layout lane N=l%16, VGPR v -> M = v + 8g
#pragma unroll
  for (int mi = 0; mi < 2; ++mi) {
#pragma unroll
    for (int ni = 0; ni < 2; ++ni) {
      v8f acc = (mi == 0) ? (ni == 0 ? acc00 : acc01) : (ni == 0 ? acc10 : acc11);
#pragma unroll
      for (int v = 0; v < 8; ++v) {
        int row = rowBlk + wm * 32 + mi * 16 + v + 8 * g;
        int col = colBlk + wn * 32 + ni * 16 + lane16;
        float x0 = acc[v];
        if (bias)  x0 += bias[col];
        if (resid) x0 += resid[(size_t)row * ldc + col];
        if (outF) outF[(size_t)row * ldc + col] = x0;
        else      outH[(size_t)row * ldc + col] = (_Float16)x0;
      }
    }
  }
}

// ---------------- flash attention: wave = one 16-query tile ----------------
__global__ __launch_bounds__(256) void attn_kernel(
    const _Float16* __restrict__ qkv, const _Float16* __restrict__ vt,
    _Float16* __restrict__ z)
{
  __shared__ __align__(16) _Float16 Plds[8][16][40]; // per-wave P tile 16x32 (+pad)
  const int tid = threadIdx.x;
  const int wave = tid >> 5, lane = tid & 31;
  const int lane16 = lane & 15, g = lane >> 4;
  const int qt = blockIdx.x * 8 + wave;
  const int h = blockIdx.y, b = blockIdx.z;
  const int qbase = qt * 16;

  const _Float16* qrow  = qkv + (size_t)(b * S_ + qbase + lane16) * (3 * DM_) + h * DH_;
  const _Float16* kbase = qkv + (size_t)(b * S_) * (3 * DM_) + DM_ + h * DH_;
  const _Float16* vtb   = vt + ((size_t)(b * NH_ + h)) * DH_ * S_;

  // Q fragments for the two 32-wide d-chunks (kept in registers for whole loop)
  Frag16 Qf0, Qf1;
  Qf0.h[0] = *(const h8*)(qrow + 16 * g);
  Qf0.h[1] = *(const h8*)(qrow + 16 * g + 8);
  Qf1.h[0] = *(const h8*)(qrow + 32 + 16 * g);
  Qf1.h[1] = *(const h8*)(qrow + 32 + 16 * g + 8);

  v8f O0 = {}, O1 = {}, O2 = {}, O3 = {};
  float mx[8], ls[8];
#pragma unroll
  for (int v = 0; v < 8; ++v) { mx[v] = -1e30f; ls[v] = 0.f; }
  const float scale = 0.125f; // 1/sqrt(64)
  const int nchunks = (qbase + 15) / 32 + 1;
  _Float16 (*Pw)[40] = Plds[wave];

  for (int j = 0; j < nchunks; ++j) {
    int k0 = j * 32;
    v8f sc0 = {}, sc1 = {};
    { // keys k0 .. k0+15
      const _Float16* kr = kbase + (size_t)(k0 + lane16) * (3 * DM_);
      Frag16 Kb0, Kb1;
      Kb0.h[0] = *(const h8*)(kr + 16 * g);       Kb0.h[1] = *(const h8*)(kr + 16 * g + 8);
      Kb1.h[0] = *(const h8*)(kr + 32 + 16 * g);  Kb1.h[1] = *(const h8*)(kr + 32 + 16 * g + 8);
      sc0 = wmma_f16(Qf0.v, Kb0.v, sc0);
      sc0 = wmma_f16(Qf1.v, Kb1.v, sc0);
    }
    { // keys k0+16 .. k0+31
      const _Float16* kr = kbase + (size_t)(k0 + 16 + lane16) * (3 * DM_);
      Frag16 Kb0, Kb1;
      Kb0.h[0] = *(const h8*)(kr + 16 * g);       Kb0.h[1] = *(const h8*)(kr + 16 * g + 8);
      Kb1.h[0] = *(const h8*)(kr + 32 + 16 * g);  Kb1.h[1] = *(const h8*)(kr + 32 + 16 * g + 8);
      sc1 = wmma_f16(Qf0.v, Kb0.v, sc1);
      sc1 = wmma_f16(Qf1.v, Kb1.v, sc1);
    }
    // mask + online softmax (per row v+8g; 16-lane shuffle reductions)
#pragma unroll
    for (int v = 0; v < 8; ++v) {
      int row = qbase + v + 8 * g;
      float s0 = sc0[v] * scale; if (k0 + lane16 > row)      s0 = -1e30f;
      float s1 = sc1[v] * scale; if (k0 + 16 + lane16 > row) s1 = -1e30f;
      float mv = fmaxf(s0, s1);
      mv = fmaxf(mv, __shfl_xor(mv, 1));
      mv = fmaxf(mv, __shfl_xor(mv, 2));
      mv = fmaxf(mv, __shfl_xor(mv, 4));
      mv = fmaxf(mv, __shfl_xor(mv, 8));
      float m2 = fmaxf(mx[v], mv);
      float alpha = __expf(mx[v] - m2);
      mx[v] = m2;
      float p0 = __expf(s0 - m2);
      float p1 = __expf(s1 - m2);
      float rs = p0 + p1;
      rs += __shfl_xor(rs, 1);
      rs += __shfl_xor(rs, 2);
      rs += __shfl_xor(rs, 4);
      rs += __shfl_xor(rs, 8);
      ls[v] = ls[v] * alpha + rs;
      O0[v] = O0[v] * alpha; O1[v] = O1[v] * alpha;
      O2[v] = O2[v] * alpha; O3[v] = O3[v] * alpha;
      Pw[v + 8 * g][lane16]      = (_Float16)p0;
      Pw[v + 8 * g][16 + lane16] = (_Float16)p1;
    }
    asm volatile("" ::: "memory"); // keep LDS store->load order (same-wave LDS is in-order in HW)
    // reload P as A fragment (16 queries x 32 keys)
    Frag16 Pf;
    Pf.h[0] = *(const h8*)&Pw[lane16][16 * g];
    Pf.h[1] = *(const h8*)&Pw[lane16][16 * g + 8];
    // V fragments: vt[d][s], contiguous over keys
    {
      const _Float16* vr = vtb + (size_t)lane16 * S_ + k0 + 16 * g;
      Frag16 Vf; Vf.h[0] = *(const h8*)vr; Vf.h[1] = *(const h8*)(vr + 8);
      O0 = wmma_f16(Pf.v, Vf.v, O0);
    }
    {
      const _Float16* vr = vtb + (size_t)(16 + lane16) * S_ + k0 + 16 * g;
      Frag16 Vf; Vf.h[0] = *(const h8*)vr; Vf.h[1] = *(const h8*)(vr + 8);
      O1 = wmma_f16(Pf.v, Vf.v, O1);
    }
    {
      const _Float16* vr = vtb + (size_t)(32 + lane16) * S_ + k0 + 16 * g;
      Frag16 Vf; Vf.h[0] = *(const h8*)vr; Vf.h[1] = *(const h8*)(vr + 8);
      O2 = wmma_f16(Pf.v, Vf.v, O2);
    }
    {
      const _Float16* vr = vtb + (size_t)(48 + lane16) * S_ + k0 + 16 * g;
      Frag16 Vf; Vf.h[0] = *(const h8*)vr; Vf.h[1] = *(const h8*)(vr + 8);
      O3 = wmma_f16(Pf.v, Vf.v, O3);
    }
  }
  // finalize: z[(b*S + q)][h*64 + d]
#pragma unroll
  for (int v = 0; v < 8; ++v) {
    float inv = 1.f / ls[v];
    size_t orow = (size_t)(b * S_ + qbase + v + 8 * g) * DM_ + h * DH_;
    z[orow + lane16]      = (_Float16)(O0[v] * inv);
    z[orow + 16 + lane16] = (_Float16)(O1[v] * inv);
    z[orow + 32 + lane16] = (_Float16)(O2[v] * inv);
    z[orow + 48 + lane16] = (_Float16)(O3[v] * inv);
  }
}

// ---------------- launch ----------------
extern "C" void kernel_launch(void* const* d_in, const int* in_sizes, int n_in,
                              void* d_out, int out_size, void* d_ws, size_t ws_size,
                              hipStream_t stream) {
  (void)in_sizes; (void)n_in; (void)out_size; (void)ws_size;
  const float* x     = (const float*)d_in[0];
  const float* W_Q   = (const float*)d_in[2];
  const float* b_Q   = (const float*)d_in[3];
  const float* W_K   = (const float*)d_in[4];
  const float* b_K   = (const float*)d_in[5];
  const float* W_V   = (const float*)d_in[6];
  const float* b_V   = (const float*)d_in[7];
  const float* W_O   = (const float*)d_in[8];
  const float* b_O   = (const float*)d_in[9];
  const float* ln1_w = (const float*)d_in[10];
  const float* ln1_b = (const float*)d_in[11];
  const float* ln2_w = (const float*)d_in[12];
  const float* ln2_b = (const float*)d_in[13];
  const float* W_in  = (const float*)d_in[14];
  const float* b_in  = (const float*)d_in[15];
  const float* W_out = (const float*)d_in[16];
  const float* b_out = (const float*)d_in[17];
  const float* mlw   = (const float*)d_in[18];
  const float* mlb   = (const float*)d_in[19];

  size_t off = 0;
  auto alloc = [&](size_t bytes) -> void* {
    void* p = (char*)d_ws + off;
    off += (bytes + 255) & ~(size_t)255;
    return p;
  };
  _Float16* xn_h   = (_Float16*)alloc((size_t)ROWS_ * DM_ * 2);
  _Float16* wqkv_h = (_Float16*)alloc((size_t)DM_ * 3 * DM_ * 2);
  float*    bqkv   = (float*)   alloc((size_t)3 * DM_ * 4);
  _Float16* qkv_h  = (_Float16*)alloc((size_t)ROWS_ * 3 * DM_ * 2);
  _Float16* vt_h   = (_Float16*)alloc((size_t)B_ * NH_ * DH_ * S_ * 2);
  _Float16* z_h    = (_Float16*)alloc((size_t)ROWS_ * DM_ * 2);
  _Float16* wo_h   = (_Float16*)alloc((size_t)DM_ * DM_ * 2);
  float*    rmid   = (float*)   alloc((size_t)ROWS_ * DM_ * 4);
  _Float16* xm_h   = (_Float16*)alloc((size_t)ROWS_ * DM_ * 2);
  _Float16* win_h  = (_Float16*)alloc((size_t)DM_ * DMLP_ * 2);
  float*    pre    = (float*)   alloc((size_t)ROWS_ * DMLP_ * 4);
  _Float16* post_h = (_Float16*)alloc((size_t)ROWS_ * DMLP_ * 2);
  _Float16* wout_h = (_Float16*)alloc((size_t)DMLP_ * DM_ * 2);

  // weight conversion / packing
  pack_qkv_w_kernel<<<(DM_ * 3 * DM_) / 256, 256, 0, stream>>>(W_Q, W_K, W_V, wqkv_h);
  pack_qkv_b_kernel<<<(3 * DM_ + 255) / 256, 256, 0, stream>>>(b_Q, b_K, b_V, bqkv);
  cvt_f16_kernel<<<(DM_ * DM_) / 256, 256, 0, stream>>>(W_O, wo_h, DM_ * DM_);
  cvt_f16_kernel<<<(DM_ * DMLP_) / 256, 256, 0, stream>>>(W_in, win_h, DM_ * DMLP_);
  cvt_f16_kernel<<<(DMLP_ * DM_) / 256, 256, 0, stream>>>(W_out, wout_h, DMLP_ * DM_);

  // LN1
  ln_kernel<<<ROWS_, 256, 0, stream>>>(x, ln1_w, ln1_b, xn_h, DM_);
  // fused QKV projection: [4096,1024] x [1024,3072] -> f16
  gemm_f16_kernel<<<dim3((3 * DM_) / 64, ROWS_ / 128), 256, 0, stream>>>(
      xn_h, DM_, wqkv_h, 3 * DM_, bqkv, nullptr, nullptr, qkv_h, 3 * DM_, DM_);
  // V transpose for attention B-fragments
  transpose_v_kernel<<<(B_ * NH_ * DH_ * S_) / 256, 256, 0, stream>>>(qkv_h, vt_h);
  // causal flash attention
  attn_kernel<<<dim3((S_ / 16) / 8, NH_, B_), 256, 0, stream>>>(qkv_h, vt_h, z_h);
  // O projection + residual(x) -> resid_mid (f32)
  gemm_f16_kernel<<<dim3(DM_ / 64, ROWS_ / 128), 256, 0, stream>>>(
      z_h, DM_, wo_h, DM_, b_O, x, rmid, nullptr, DM_, DM_);
  // LN2
  ln_kernel<<<ROWS_, 256, 0, stream>>>(rmid, ln2_w, ln2_b, xm_h, DM_);
  // MLP in: [4096,1024] x [1024,4096] -> pre (f32)
  gemm_f16_kernel<<<dim3(DMLP_ / 64, ROWS_ / 128), 256, 0, stream>>>(
      xm_h, DM_, win_h, DMLP_, b_in, nullptr, pre, nullptr, DMLP_, DM_);
  // SoLU + mlp LN -> post (f16)
  solu_kernel<<<ROWS_, 256, 0, stream>>>(pre, mlw, mlb, post_h, DMLP_);
  // MLP out + residual(resid_mid) -> d_out (f32)
  gemm_f16_kernel<<<dim3(DM_ / 64, ROWS_ / 128), 256, 0, stream>>>(
      post_h, DMLP_, wout_h, DM_, b_out, rmid, (float*)d_out, nullptr, DM_, DMLP_);
}
